// GATRoostLayer_53274774340080
// MI455X (gfx1250) — matrix operand order repaired and checked by reading, not compiled, not run on hardware
//
#include <hip/hip_runtime.h>
#include <hip/hip_bf16.h>

#define NN      50000      // nodes
#define NE      400000     // edges
#define DIN     64
#define DOUTF   64
#define GH      128        // msg hidden
#define FH      64         // gate hidden
#define HEADS   4
#define NEG     0.2f

typedef __attribute__((ext_vector_type(16))) __bf16        v16bf;
typedef __attribute__((ext_vector_type(8)))  float         v8f;
typedef __attribute__((ext_vector_type(8)))  unsigned int  v8u;

// ---- workspace layout (bytes) ----
#define OFF_ROWSUM  0u                              // H*NN f32   = 800 KB
#define OFF_G1FRAG  (1u<<20)                        // 64 KB
#define OFF_M1FRAG  ((1u<<20) + 65536u)             // 128 KB
#define OFF_M2FRAG  ((1u<<20) + 65536u + 131072u)   // 64 KB
#define OFF_AEDGE   (2u<<20)                        // H*NE f32   = 6.4 MB
#define OFF_XBF     (9u<<20)                        // NN*64 bf16 = 6.4 MB
// total < 16 MB

__device__ __forceinline__ float lrelu(float v) { return v >= 0.0f ? v : NEG * v; }

__device__ __forceinline__ unsigned short f2bf(float f) {
    unsigned int u = __builtin_bit_cast(unsigned int, f);
    u += 0x7FFFu + ((u >> 16) & 1u);          // round-to-nearest-even
    return (unsigned short)(u >> 16);
}

__device__ __forceinline__ v8f wmma_bf16(v16bf a, v16bf b, v8f c) {
    return __builtin_amdgcn_wmma_f32_16x16x32_bf16(false, a, false, b, (short)0, c, false, false);
}

// A-matrix 16x32 bf16 fragment from an LDS tile stored row-major [16][128].
// Per ISA: lane L holds row m=L&15; VGPR v packs K = (v>=4?16:0) + (L>>4)*8 + 2*(v&3) (+1).
__device__ __forceinline__ v16bf load_a_frag(const unsigned short* lds, int lane, int kbase) {
    const int m = lane & 15, half = lane >> 4;
    v8u au;
#pragma unroll
    for (int v = 0; v < 8; ++v) {
        int k = kbase + ((v & 4) ? 16 : 0) + half * 8 + ((v & 3) << 1);
        au[v] = *(const unsigned int*)(lds + m * 128 + k);
    }
    return __builtin_bit_cast(v16bf, au);
}

// B fragments are pre-packed lane-major: 16 contiguous bf16 per lane (512 per frag).
__device__ __forceinline__ v16bf load_b_frag(const unsigned short* frag, int lane) {
    return __builtin_bit_cast(v16bf, *(const v8u*)(frag + lane * 16));
}

// ---------------- prep kernels ----------------
__global__ void zero_kernel(float* rowsum, int n0, float* out, int n1) {
    int i = blockIdx.x * blockDim.x + threadIdx.x;
    int stride = gridDim.x * blockDim.x;
    for (int j = i; j < n0; j += stride) rowsum[j] = 0.0f;
    for (int j = i; j < n1; j += stride) out[j]    = 0.0f;
}

__global__ void xbf_kernel(const float* __restrict__ x, unsigned short* __restrict__ xbf) {
    int i = blockIdx.x * blockDim.x + threadIdx.x;
    if (i < NN * DIN) xbf[i] = f2bf(x[i]);
}

// pack weight W[h][g][f] (f contiguous, f-dim = 128) into B-frag layout.
// B element (k,n) of frag (h,nt,kk) = W[h][nt*16+n][kk*32+k].
// lane L supplies n=L&15, K = (L>>4)*8 + j (j<8), 16 + (L>>4)*8 + (j-8) (j>=8).
__global__ void prep_frags(const float* __restrict__ w, unsigned short* __restrict__ frag, int NT) {
    int t = blockIdx.x * blockDim.x + threadIdx.x;
    int total = HEADS * NT * 4 * 32;
    if (t >= total) return;
    int lane = t & 31, fid = t >> 5;
    int kk = fid & 3, nt = (fid >> 2) % NT, h = fid / (4 * NT);
    const float* src = w + (((long long)h * NT * 16) + nt * 16 + (lane & 15)) * 128
                         + kk * 32 + (lane >> 4) * 8;
    unsigned short* dst = frag + (long long)fid * 512 + lane * 16;
#pragma unroll
    for (int j = 0; j < 16; ++j) dst[j] = f2bf(src[j + (j & 8)]);
}

// load one 16-edge tile's features (bf16) + row/col ids into per-wave LDS
__device__ __forceinline__ void fill_edge_tile(int tile, const long long* eidx,
                                               const unsigned short* xbf,
                                               unsigned short* eL, int* rowL, int* colL, int lane) {
    int m = lane >> 1, hh = lane & 1;
    long long e = (long long)tile * 16 + m;
    long long node = eidx[(long long)hh * NE + e];
    if (hh == 0) rowL[m] = (int)node; else colL[m] = (int)node;
    const uint4* src = (const uint4*)(xbf + node * 64);
    uint4* dst = (uint4*)(eL + m * 128 + hh * 64);
#pragma unroll
    for (int j = 0; j < 8; ++j) dst[j] = src[j];
}

// ---------------- pass 1: gate MLP + exp + rowsum ----------------
__global__ void __launch_bounds__(128)
gate_kernel(const long long* __restrict__ eidx, const unsigned short* __restrict__ xbf,
            const float* __restrict__ pos, const unsigned short* __restrict__ g1f,
            const float* __restrict__ gw2, float* __restrict__ aedge, float* __restrict__ rowsum) {
    __shared__ __align__(16) unsigned short e_s[4][16 * 128];
    __shared__ int row_s[4][16], col_s[4][16];
    const int wave = threadIdx.x >> 5, lane = threadIdx.x & 31;
    const int tile = blockIdx.x * 4 + wave;
    const int half = lane >> 4;

    fill_edge_tile(tile, eidx, xbf, e_s[wave], row_s[wave], col_s[wave], lane);
    __syncthreads();

    v16bf ae[4];
#pragma unroll
    for (int kk = 0; kk < 4; ++kk) ae[kk] = load_a_frag(e_s[wave], lane, kk * 32);

    for (int h = 0; h < HEADS; ++h) {
        float part[8];
#pragma unroll
        for (int r = 0; r < 8; ++r) part[r] = 0.0f;
#pragma unroll
        for (int nt = 0; nt < 4; ++nt) {
            v8f c = {};
#pragma unroll
            for (int kk = 0; kk < 4; ++kk)
                c = wmma_bf16(ae[kk], load_b_frag(g1f + ((h * 4 + nt) * 4 + kk) * 512, lane), c);
            float w2v = gw2[h * FH + nt * 16 + (lane & 15)];
#pragma unroll
            for (int r = 0; r < 8; ++r) part[r] += lrelu(c[r]) * w2v;
        }
        // reduce over 16 lanes of each half (they hold n=0..15 of the same edges)
#pragma unroll
        for (int r = 0; r < 8; ++r) {
#pragma unroll
            for (int mask = 1; mask < 16; mask <<= 1)
                part[r] += __shfl_xor(part[r], mask, 32);
            part[r] = lrelu(part[r]);
        }
#pragma unroll
        for (int r = 0; r < 8; ++r) {
            if ((lane & 15) == r) {
                int m = half * 8 + r;
                int e = tile * 16 + m;
                float av = pos[col_s[wave][m]] * __expf(part[r]);
                aedge[(long long)h * NE + e] = av;
                atomicAdd(&rowsum[h * NN + row_s[wave][m]], av);
            }
        }
    }
}

// ---------------- pass 2: message MLP, normalize, scatter ----------------
__global__ void __launch_bounds__(128)
msg_kernel(const long long* __restrict__ eidx, const unsigned short* __restrict__ xbf,
           const unsigned short* __restrict__ m1f, const unsigned short* __restrict__ m2f,
           const float* __restrict__ aedge, const float* __restrict__ rowsum,
           float* __restrict__ out) {
    __shared__ __align__(16) unsigned short e_s[4][16 * 128];
    __shared__ __align__(16) unsigned short m1_s[4][16 * 128];
    __shared__ int row_s[4][16], col_s[4][16];
    const int wave = threadIdx.x >> 5, lane = threadIdx.x & 31;
    const int tile = blockIdx.x * 4 + wave;
    const int half = lane >> 4;

    fill_edge_tile(tile, eidx, xbf, e_s[wave], row_s[wave], col_s[wave], lane);
    __syncthreads();

    v16bf ae[4];
#pragma unroll
    for (int kk = 0; kk < 4; ++kk) ae[kk] = load_a_frag(e_s[wave], lane, kk * 32);

    v8f acc[4];
#pragma unroll
    for (int bt = 0; bt < 4; ++bt) acc[bt] = (v8f){};

    for (int h = 0; h < HEADS; ++h) {
        // m1 = lrelu(e @ msg_w1^T) : [16 x 128], written to LDS as next A operand
#pragma unroll
        for (int nt = 0; nt < 8; ++nt) {
            v8f c = {};
#pragma unroll
            for (int kk = 0; kk < 4; ++kk)
                c = wmma_bf16(ae[kk], load_b_frag(m1f + ((h * 8 + nt) * 4 + kk) * 512, lane), c);
#pragma unroll
            for (int r = 0; r < 8; ++r)
                m1_s[wave][(half * 8 + r) * 128 + nt * 16 + (lane & 15)] = f2bf(lrelu(c[r]));
        }
        __syncthreads();

        v16bf am[4];
#pragma unroll
        for (int kk = 0; kk < 4; ++kk) am[kk] = load_a_frag(m1_s[wave], lane, kk * 32);

        float an[8];
#pragma unroll
        for (int r = 0; r < 8; ++r) {
            int m = half * 8 + r;
            float a  = aedge[(long long)h * NE + tile * 16 + m];
            float rs = rowsum[h * NN + row_s[wave][m]];
            an[r] = a * (1.0f / rs + 1e-16f);
        }
        // beta = lrelu(m1 @ msg_w2^T) : [16 x 64]
#pragma unroll
        for (int bt = 0; bt < 4; ++bt) {
            v8f cb = {};
#pragma unroll
            for (int kk = 0; kk < 4; ++kk)
                cb = wmma_bf16(am[kk], load_b_frag(m2f + ((h * 4 + bt) * 4 + kk) * 512, lane), cb);
#pragma unroll
            for (int r = 0; r < 8; ++r) acc[bt][r] += an[r] * lrelu(cb[r]);
        }
        __syncthreads();
    }
#pragma unroll
    for (int bt = 0; bt < 4; ++bt)
#pragma unroll
        for (int r = 0; r < 8; ++r) {
            int node = row_s[wave][half * 8 + r];
            atomicAdd(&out[(long long)node * DOUTF + bt * 16 + (lane & 15)], 0.25f * acc[bt][r]);
        }
}

extern "C" void kernel_launch(void* const* d_in, const int* in_sizes, int n_in,
                              void* d_out, int out_size, void* d_ws, size_t ws_size,
                              hipStream_t stream) {
    const float*     x    = (const float*)d_in[0];
    const long long* eidx = (const long long*)d_in[1];
    const float*     pos  = (const float*)d_in[2];
    const float*     gw1  = (const float*)d_in[3];
    const float*     gw2  = (const float*)d_in[4];
    const float*     mw1  = (const float*)d_in[5];
    const float*     mw2  = (const float*)d_in[6];
    float* out = (float*)d_out;
    char*  ws  = (char*)d_ws;

    float*          rowsum = (float*)(ws + OFF_ROWSUM);
    unsigned short* g1f    = (unsigned short*)(ws + OFF_G1FRAG);
    unsigned short* m1f    = (unsigned short*)(ws + OFF_M1FRAG);
    unsigned short* m2f    = (unsigned short*)(ws + OFF_M2FRAG);
    float*          aedge  = (float*)(ws + OFF_AEDGE);
    unsigned short* xbf    = (unsigned short*)(ws + OFF_XBF);

    zero_kernel<<<1024, 256, 0, stream>>>(rowsum, HEADS * NN, out, NN * DOUTF);
    xbf_kernel<<<(NN * DIN + 255) / 256, 256, 0, stream>>>(x, xbf);
    prep_frags<<<8,  256, 0, stream>>>(gw1, g1f, 4);  // [H,64,128]
    prep_frags<<<16, 256, 0, stream>>>(mw1, m1f, 8);  // [H,128,128]
    prep_frags<<<8,  256, 0, stream>>>(mw2, m2f, 4);  // [H,64,128]

    const int nblk = (NE / 16) / 4;   // 25000 tiles, 4 waves/block
    gate_kernel<<<nblk, 128, 0, stream>>>(eidx, xbf, pos, g1f, gw2, aedge, rowsum);
    msg_kernel <<<nblk, 128, 0, stream>>>(eidx, xbf, m1f, m2f, aedge, rowsum, out);
}
